// MultiHeadSelfAttention_74955769250219
// MI455X (gfx1250) — matrix operand order repaired and checked by reading, not compiled, not run on hardware
//
#include <hip/hip_runtime.h>
#include <hip/hip_bf16.h>

// ---------------------------------------------------------------------------
// MI455X (gfx1250) multi-head self-attention, bf16 WMMA pipeline.
// D_MODEL=1024, NUM_HEADS=16, HEAD_DIM=64, B=4, T=2048.
//
// Data path: weights pre-converted to bf16 AND pre-transposed (K-contiguous
// per output column) so GEMM tile staging is a pure copy -> async global->LDS
// (GLOBAL_LOAD_ASYNC_TO_LDS_B128, ASYNCcnt) where the toolchain exposes it.
// All matrix math via v_wmma_f32_16x16x32_bf16.
// ---------------------------------------------------------------------------

typedef __attribute__((ext_vector_type(16))) __bf16         bf16x16;
typedef __attribute__((ext_vector_type(16))) unsigned short u16x16;
typedef __attribute__((ext_vector_type(8)))  float          f32x8;
typedef int v4i32 __attribute__((vector_size(16)));   // matches builtin signature

struct alignas(4) US2 { unsigned short x, y; };

#define DMODEL 1024
#define NHEAD  16
#define HDIM   64
#define SEQ    2048
#define BATCH  4
#define MTOT   (BATCH * SEQ)      // 8192 rows for the projections

#define BK     32
#define LDS_A1 34                 // padded stride for converted-x tile (b32 stores)
#define LDS_B  40                 // 80B rows: 16B-aligned + conflict-free mod 256B

#if defined(__has_builtin)
#if __has_builtin(__builtin_amdgcn_global_load_async_to_lds_b128)
#define HAVE_ASYNC_COPY 1
#endif
#endif

#ifdef HAVE_ASYNC_COPY
#define AS1 __attribute__((address_space(1)))
#define AS3 __attribute__((address_space(3)))
static __device__ __forceinline__ void async_copy16(const unsigned short* g,
                                                    unsigned short* l) {
    __builtin_amdgcn_global_load_async_to_lds_b128(
        (AS1 v4i32*)(v4i32*)g,    // global source (AS1, v4i pointee per builtin)
        (AS3 v4i32*)(v4i32*)l,    // LDS destination (AS3)
        0, 0);
}
#define ASYNC_WAIT() asm volatile("s_wait_asynccnt 0x0" ::: "memory")
#else
static __device__ __forceinline__ void async_copy16(const unsigned short* g,
                                                    unsigned short* l) {
#pragma unroll
    for (int u = 0; u < 4; ++u)
        *(unsigned int*)(l + 2 * u) = *(const unsigned int*)(g + 2 * u);
}
#define ASYNC_WAIT() ((void)0)
#endif

static __device__ __forceinline__ unsigned short f2bf(float f) {
    unsigned int u = __float_as_uint(f);
    u += 0x7FFFu + ((u >> 16) & 1u);            // round-to-nearest-even
    return (unsigned short)(u >> 16);
}

static __device__ __forceinline__ f32x8 wmma_bf16(bf16x16 a, bf16x16 b, f32x8 c) {
    return __builtin_amdgcn_wmma_f32_16x16x32_bf16(
        /*neg_a=*/false, a, /*neg_b=*/false, b,
        /*c_mod=*/(short)0, c, /*reuse_a=*/false, /*reuse_b=*/false);
}

// A operand (16x32 bf16): lane m = lane&15, K-half = lane>>4.
// VGPR j holds K pairs: 2j + 16*(lane>>4).
static __device__ __forceinline__ bf16x16
load_a_frag(const unsigned short* base, int stride, int lane) {
    int m = lane & 15, kh = lane >> 4;
    u16x16 t;
#pragma unroll
    for (int j = 0; j < 8; ++j) {
        int k = 2 * j + 16 * kh;
        US2 p = *(const US2*)(base + m * stride + k);
        t[2 * j] = p.x; t[2 * j + 1] = p.y;
    }
    return __builtin_bit_cast(bf16x16, t);
}

// B operand (32x16 bf16) from K-contiguous ("transposed") storage [n][k]:
// lane n = lane&15, K-group = lane>>4; VGPR j holds K = 2j+16*(lane>>4), +1.
static __device__ __forceinline__ bf16x16
load_bT_frag(const unsigned short* base, int stride, int lane) {
    int n = lane & 15, kg = lane >> 4;
    u16x16 t;
#pragma unroll
    for (int j = 0; j < 8; ++j) {
        int k = 2 * j + 16 * kg;
        US2 p = *(const US2*)(base + n * stride + k);
        t[2 * j] = p.x; t[2 * j + 1] = p.y;
    }
    return __builtin_bit_cast(bf16x16, t);
}

// B operand (32x16 bf16) from row-major [k][n] storage (used for V).
static __device__ __forceinline__ bf16x16
load_b_row_frag(const unsigned short* base, int stride, int lane) {
    int n = lane & 15, kg = lane >> 4;
    u16x16 t;
#pragma unroll
    for (int j = 0; j < 8; ++j) {
        int k = 2 * j + 16 * kg;
        t[2 * j]     = base[(k)     * stride + n];
        t[2 * j + 1] = base[(k + 1) * stride + n];
    }
    return __builtin_bit_cast(bf16x16, t);
}

// ---------------------------------------------------------------------------
// Kernel 0: convert weights f32 -> bf16 AND transpose to [n][k] (K-contiguous
// per output column). One-shot, bandwidth bound.
// ---------------------------------------------------------------------------
__global__ __launch_bounds__(256) void convert_weights_kernel(
    const float* __restrict__ Wqkv, const float* __restrict__ Wout,
    unsigned short* __restrict__ wqT, unsigned short* __restrict__ woT) {
    const int NQ = DMODEL * 3 * DMODEL;            // 3,145,728
    int i = blockIdx.x * 256 + threadIdx.x;
    if (i < NQ) {
        int k = i / (3 * DMODEL), n = i % (3 * DMODEL);
        wqT[(size_t)n * DMODEL + k] = f2bf(Wqkv[i]);
    } else {
        int j = i - NQ;                             // < 1,048,576
        int k = j / DMODEL, n = j % DMODEL;
        woT[(size_t)n * DMODEL + k] = f2bf(Wout[j]);
    }
}

// ---------------------------------------------------------------------------
// Kernel 1: QKV projection.  qkv = x @ W_qkv + b_qkv, scatter to Q/K/V in
// (B,H,T,hd) bf16 layout; Q pre-scaled by 1/sqrt(hd).
// Block = 256 threads (8 waves); tile 64(M) x 128(N) x 32(K).
// A: f32->bf16 convert, packed b32 LDS stores.  B: async b128 tile copy.
// ---------------------------------------------------------------------------
__global__ __launch_bounds__(256) void qkv_gemm_kernel(
    const float* __restrict__ x, const unsigned short* __restrict__ WbT,
    const float* __restrict__ bqkv,
    unsigned short* __restrict__ Q, unsigned short* __restrict__ K,
    unsigned short* __restrict__ V) {
    __shared__ unsigned short As[64 * LDS_A1];
    __shared__ unsigned short Bs[128 * LDS_B];   // [n][k], K-contiguous

    const int tid = threadIdx.x, lane = tid & 31, w = tid >> 5;
    const int m0 = blockIdx.x * 64;
    const int n0 = blockIdx.y * 128;
    const int m_off = (w & 3) * 16;            // 4 waves along M
    const int n_off = (w >> 2) * 64;           // 2 wave groups along N

    f32x8 acc[4] = {};

    for (int kt = 0; kt < DMODEL / BK; ++kt) {
        const int k0 = kt * BK;
        __syncthreads();
        // Stage B via async b128 copies: 128 rows x 64B = 512 chunks (2/thread).
#pragma unroll
        for (int i = 0; i < 2; ++i) {
            int id = tid + i * 256;
            int n = id >> 2, c = (id & 3) * 8;
            async_copy16(WbT + (size_t)(n0 + n) * DMODEL + k0 + c,
                         &Bs[n * LDS_B + c]);
        }
        // Stage A: 64x32 f32 -> bf16, packed pairs, b32 stores (4/thread).
#pragma unroll
        for (int i = 0; i < 4; ++i) {
            int e = tid + i * 256;
            int r = e >> 4, c = (e & 15) * 2;
            const float* src = &x[(size_t)(m0 + r) * DMODEL + k0 + c];
            unsigned int pk = (unsigned int)f2bf(src[0]) |
                              ((unsigned int)f2bf(src[1]) << 16);
            *(unsigned int*)&As[r * LDS_A1 + c] = pk;
        }
        if (kt + 1 < DMODEL / BK) {   // hint next tiles toward L2
            __builtin_prefetch(&x[(size_t)(m0 + (tid >> 2)) * DMODEL + k0 + BK], 0, 1);
            __builtin_prefetch(&WbT[(size_t)(n0 + (tid >> 1)) * DMODEL + k0 + BK], 0, 1);
        }
        ASYNC_WAIT();
        __syncthreads();

        bf16x16 a = load_a_frag(&As[m_off * LDS_A1], LDS_A1, lane);
#pragma unroll
        for (int nt = 0; nt < 4; ++nt) {
            bf16x16 b = load_bT_frag(&Bs[(n_off + nt * 16) * LDS_B], LDS_B, lane);
            acc[nt] = wmma_bf16(a, b, acc[nt]);
        }
    }

    // Epilogue: bias, scale Q, scatter to (B,H,T,hd) bf16.
    const int mrow_base = m0 + m_off + 8 * (lane >> 4);
#pragma unroll
    for (int nt = 0; nt < 4; ++nt) {
        int ng = n0 + n_off + nt * 16 + (lane & 15);
        float bias = bqkv[ng];
        int which = ng >> 10;          // 0=Q 1=K 2=V
        int dloc = ng & 1023;
        int h = dloc >> 6, dd = dloc & 63;
        unsigned short* dst = (which == 0) ? Q : (which == 1) ? K : V;
        float scl = (which == 0) ? 0.125f : 1.0f;   // 1/sqrt(64)
#pragma unroll
        for (int r = 0; r < 8; ++r) {
            int mg = mrow_base + r;
            int bb = mg >> 11, tt = mg & (SEQ - 1);
            float v = (acc[nt][r] + bias) * scl;
            dst[(((size_t)bb * NHEAD + h) * SEQ + tt) * HDIM + dd] = f2bf(v);
        }
    }
}

// ---------------------------------------------------------------------------
// Kernel 2: causal flash attention, bf16 WMMA, online softmax.
// Block = 128 threads (4 waves); wave owns 16 query rows; key tiles of 32.
// K/V are L2-resident (32 MB << 192 MB L2) -> direct global fragment loads.
// ---------------------------------------------------------------------------
__global__ __launch_bounds__(128) void flash_attn_kernel(
    const unsigned short* __restrict__ Q, const unsigned short* __restrict__ K,
    const unsigned short* __restrict__ V, const unsigned char* __restrict__ mask,
    unsigned short* __restrict__ AO) {
    __shared__ unsigned short Ps[4 * 16 * 34];   // per-wave P re-fragment buffer

    const int lane = threadIdx.x & 31, w = threadIdx.x >> 5;
    unsigned short* pbuf = &Ps[w * 16 * 34];
    const int b = blockIdx.x >> 4, h = blockIdx.x & (NHEAD - 1);
    const int q0 = blockIdx.y * 64 + w * 16;

    const size_t bh = (size_t)(b * NHEAD + h) * SEQ * HDIM;
    const unsigned short* Qb = Q + bh;
    const unsigned short* Kb = K + bh;
    const unsigned short* Vb = V + bh;
    const unsigned char* mb = mask + (size_t)b * SEQ;

    const float NEG_INF = -__builtin_inff();
    const int halfsel = lane >> 4, col = lane & 15;

    bf16x16 qf0 = load_a_frag(Qb + (size_t)q0 * HDIM + 0,  HDIM, lane);
    bf16x16 qf1 = load_a_frag(Qb + (size_t)q0 * HDIM + 32, HDIM, lane);

    f32x8 o[4] = {};
    float mrun[8], lrun[8];
#pragma unroll
    for (int r = 0; r < 8; ++r) { mrun[r] = NEG_INF; lrun[r] = 0.f; }

    const int nkt = (q0 + 16 + 31) >> 5;          // causal key-tile bound
    for (int kt = 0; kt < nkt; ++kt) {
        const int k0 = kt * 32;
        // S = Q @ K^T for 16 rows x 32 keys (two 16-key subtiles, hd split 2x32)
        f32x8 s0 = {}, s1 = {};
        s0 = wmma_bf16(qf0, load_bT_frag(Kb + (size_t)(k0)      * HDIM + 0,  HDIM, lane), s0);
        s0 = wmma_bf16(qf1, load_bT_frag(Kb + (size_t)(k0)      * HDIM + 32, HDIM, lane), s0);
        s1 = wmma_bf16(qf0, load_bT_frag(Kb + (size_t)(k0 + 16) * HDIM + 0,  HDIM, lane), s1);
        s1 = wmma_bf16(qf1, load_bT_frag(Kb + (size_t)(k0 + 16) * HDIM + 32, HDIM, lane), s1);

        // Causal + padding mask.
        const int kA = k0 + col, kB = kA + 16;
        const bool mA = mb[kA] != 0, mB = mb[kB] != 0;
#pragma unroll
        for (int r = 0; r < 8; ++r) {
            int qrow = q0 + r + 8 * halfsel;
            if (kA > qrow || !mA) s0[r] = NEG_INF;
            if (kB > qrow || !mB) s1[r] = NEG_INF;
        }

        // Online softmax: each row lives in one 16-lane half (C layout).
#pragma unroll
        for (int r = 0; r < 8; ++r) {
            float mt = fmaxf(s0[r], s1[r]);
            mt = fmaxf(mt, __shfl_xor(mt, 1, 32));
            mt = fmaxf(mt, __shfl_xor(mt, 2, 32));
            mt = fmaxf(mt, __shfl_xor(mt, 4, 32));
            mt = fmaxf(mt, __shfl_xor(mt, 8, 32));
            float mnew = fmaxf(mrun[r], mt);
            float corr = (mrun[r] == NEG_INF) ? 0.f : __expf(mrun[r] - mnew);
            float p0 = (mnew == NEG_INF) ? 0.f : __expf(s0[r] - mnew);
            float p1 = (mnew == NEG_INF) ? 0.f : __expf(s1[r] - mnew);
            float rs = p0 + p1;
            rs += __shfl_xor(rs, 1, 32);
            rs += __shfl_xor(rs, 2, 32);
            rs += __shfl_xor(rs, 4, 32);
            rs += __shfl_xor(rs, 8, 32);
            lrun[r] = lrun[r] * corr + rs;
            mrun[r] = mnew;
#pragma unroll
            for (int dt = 0; dt < 4; ++dt) o[dt][r] *= corr;
            // C-layout -> LDS (row-major 16x32) for re-fragmentation as A.
            pbuf[(r + 8 * halfsel) * 34 + col]      = f2bf(p0);
            pbuf[(r + 8 * halfsel) * 34 + 16 + col] = f2bf(p1);
        }

        // O += P @ V  (K=32 keys, N split into 4 x 16 head-dim tiles)
        bf16x16 pf = load_a_frag(pbuf, 34, lane);
#pragma unroll
        for (int dt = 0; dt < 4; ++dt) {
            bf16x16 vf = load_b_row_frag(Vb + (size_t)k0 * HDIM + dt * 16, HDIM, lane);
            o[dt] = wmma_bf16(pf, vf, o[dt]);
        }
    }

    // Finalize: O /= l, write attn output as (B,T,H,hd) bf16 == row-major (M,1024).
#pragma unroll
    for (int r = 0; r < 8; ++r) {
        float inv = (lrun[r] > 0.f) ? 1.0f / lrun[r] : 0.f;
        int t = q0 + r + 8 * halfsel;
        size_t base = (((size_t)b * SEQ + t) * NHEAD + h) * HDIM;
#pragma unroll
        for (int dt = 0; dt < 4; ++dt)
            AO[base + dt * 16 + col] = f2bf(o[dt][r] * inv);
    }
}

// ---------------------------------------------------------------------------
// Kernel 3: output projection.  out = AO @ W_out + b_out (f32 result).
// Both A and B tiles are pure bf16 copies -> async b128 staging.
// ---------------------------------------------------------------------------
__global__ __launch_bounds__(256) void out_gemm_kernel(
    const unsigned short* __restrict__ A, const unsigned short* __restrict__ WbT,
    const float* __restrict__ bout, float* __restrict__ out) {
    __shared__ unsigned short As[64 * LDS_B];
    __shared__ unsigned short Bs[128 * LDS_B];

    const int tid = threadIdx.x, lane = tid & 31, w = tid >> 5;
    const int m0 = blockIdx.x * 64;
    const int n0 = blockIdx.y * 128;
    const int m_off = (w & 3) * 16;
    const int n_off = (w >> 2) * 64;

    f32x8 acc[4] = {};

    for (int kt = 0; kt < DMODEL / BK; ++kt) {
        const int k0 = kt * BK;
        __syncthreads();
        // Stage A: 64 rows x 64B = 256 chunks (1/thread), async.
        {
            int r = tid >> 2, c = (tid & 3) * 8;
            async_copy16(A + (size_t)(m0 + r) * DMODEL + k0 + c,
                         &As[r * LDS_B + c]);
        }
        // Stage B: 128 rows x 64B = 512 chunks (2/thread), async.
#pragma unroll
        for (int i = 0; i < 2; ++i) {
            int id = tid + i * 256;
            int n = id >> 2, c = (id & 3) * 8;
            async_copy16(WbT + (size_t)(n0 + n) * DMODEL + k0 + c,
                         &Bs[n * LDS_B + c]);
        }
        ASYNC_WAIT();
        __syncthreads();

        bf16x16 a = load_a_frag(&As[m_off * LDS_B], LDS_B, lane);
#pragma unroll
        for (int nt = 0; nt < 4; ++nt) {
            bf16x16 b = load_bT_frag(&Bs[(n_off + nt * 16) * LDS_B], LDS_B, lane);
            acc[nt] = wmma_bf16(a, b, acc[nt]);
        }
    }

    const int mrow_base = m0 + m_off + 8 * (lane >> 4);
#pragma unroll
    for (int nt = 0; nt < 4; ++nt) {
        int ng = n0 + n_off + nt * 16 + (lane & 15);
        float bias = bout[ng];
#pragma unroll
        for (int r = 0; r < 8; ++r) {
            int mg = mrow_base + r;
            out[(size_t)mg * DMODEL + ng] = acc[nt][r] + bias;
        }
    }
}

// ---------------------------------------------------------------------------
// Host launcher.
// ---------------------------------------------------------------------------
extern "C" void kernel_launch(void* const* d_in, const int* in_sizes, int n_in,
                              void* d_out, int out_size, void* d_ws, size_t ws_size,
                              hipStream_t stream) {
    (void)in_sizes; (void)n_in; (void)out_size; (void)ws_size;

    const float*         x    = (const float*)d_in[0];
    const unsigned char* amsk = (const unsigned char*)d_in[1];   // jnp.bool_
    const float*         Wqkv = (const float*)d_in[2];
    const float*         bqkv = (const float*)d_in[3];
    const float*         Wout = (const float*)d_in[4];
    const float*         bout = (const float*)d_in[5];
    float*               out  = (float*)d_out;

    // Workspace layout (ushort elements): total ~75.5 MB.
    unsigned short* ws = (unsigned short*)d_ws;
    unsigned short* wqT = ws;                                     // 3,145,728
    unsigned short* woT = wqT + (size_t)DMODEL * 3 * DMODEL;      // 1,048,576
    unsigned short* Qws = woT + (size_t)DMODEL * DMODEL;          // 8,388,608 each
    unsigned short* Kws = Qws + (size_t)MTOT * DMODEL;
    unsigned short* Vws = Kws + (size_t)MTOT * DMODEL;
    unsigned short* AOws = Vws + (size_t)MTOT * DMODEL;

    // 0) weight convert + transpose: (3*D*D + D*D) / 256 = 16384 blocks
    convert_weights_kernel<<<16384, 256, 0, stream>>>(Wqkv, Wout, wqT, woT);

    // 1) QKV projection: M=8192 (128 tiles of 64), N=3072 (24 tiles of 128)
    qkv_gemm_kernel<<<dim3(128, 24), 256, 0, stream>>>(x, wqT, bqkv, Qws, Kws, Vws);

    // 2) flash attention: (B*H = 64) x (T/64 = 32) blocks of 128 threads
    flash_attn_kernel<<<dim3(64, 32), 128, 0, stream>>>(Qws, Kws, Vws, amsk, AOws);

    // 3) output projection: M=8192, N=1024 (8 tiles of 128)
    out_gemm_kernel<<<dim3(128, 8), 256, 0, stream>>>(AOws, woT, bout, out);
}
